// MHSA_77300821394136
// MI455X (gfx1250) — matrix operand order repaired and checked by reading, not compiled, not run on hardware
//
#include <hip/hip_runtime.h>
#include <hip/hip_bf16.h>

typedef __attribute__((ext_vector_type(16))) _Float16 v16h;
typedef __attribute__((ext_vector_type(8)))  _Float16 v8h;
typedef __attribute__((ext_vector_type(8)))  float    v8f;

#define EMBED 1024
#define HEADS 16
#define HD    64
#define BATCH 2
#define SEQ   2048
#define ROWS  (BATCH*SEQ)     /* 4096 */
#define SCALE_INV 8.0f        /* scores = qk / (hd^-0.5) = qk * sqrt(64) */

#define OUT_QK  0
#define OUT_VT  1
#define OUT_F32 2

#define WMMA_F16(a,b,c) __builtin_amdgcn_wmma_f32_16x16x32_f16(false,(a),false,(b),(short)0,(c),false,false)

// ---------------------------------------------------------------------------
// Async global -> LDS copy, 16B per lane (CDNA5, tracked by ASYNCcnt).
// ISA 10/15.18: LDS destination offset comes from the VDST VGPR; generic LDS ptr low 32 bits
// are the LDS byte offset (aperture mapping, ISA 10.2).
__device__ __forceinline__ unsigned lds_off32(const void* p){
    return (unsigned)(unsigned long long)p;
}
__device__ __forceinline__ void async_ld_b128(unsigned lds_off, const void* g){
    asm volatile("global_load_async_to_lds_b128 %0, %1, off"
                 :: "v"(lds_off), "v"((unsigned long long)g) : "memory");
}

// A-operand (16x32 f16, MxK) from row-major matrix (ld elements/row).
// lane L: M=L%16, h=L/16; e 0..15 -> K=k0+16h+e
__device__ __forceinline__ v16h load_a16(const _Float16* p, int ld, int row0, int k0, int lane){
    int m = lane & 15, h = lane >> 4;
    const _Float16* r = p + (size_t)(row0 + m) * (size_t)ld;
    v8h lo = *(const v8h*)(r + k0 + 16*h);
    v8h hi = *(const v8h*)(r + k0 + 16*h + 8);
    return __builtin_shufflevector(lo, hi, 0,1,2,3,4,5,6,7,8,9,10,11,12,13,14,15);
}

// ---------------------------------------------------------------------------
__global__ __launch_bounds__(256) void cvt_f32_f16(const float* __restrict__ in,
                                                   _Float16* __restrict__ out, int n){
    int i = blockIdx.x * blockDim.x + threadIdx.x;
    if (i < n) out[i] = (_Float16)in[i];
}

// ---------------------------------------------------------------------------
// Y = A(f16, ROWS x EMBED) @ W^T(f16) + bias. Block: 128 rows x 64 cols,
// 8 waves. B-panel (64x32 f16 = 4KB) async-staged to LDS, double buffered.
// All B-tile LDS reads issued before the WMMA group (one DS drain per panel).
__global__ __launch_bounds__(256) void gemm_xWt(const _Float16* __restrict__ A,
                                                const _Float16* __restrict__ W,
                                                const float* __restrict__ bias,
                                                void* __restrict__ out, int outmode){
    __shared__ _Float16 bsm[2][64*32];           // 2 x 4KB
    int t    = threadIdx.x;
    int lane = t & 31;
    int wave = t >> 5;
    int row0 = (blockIdx.x >> 4) * 128 + wave * 16;   // 32 row-blocks
    int col0 = (blockIdx.x & 15) * 64;                // 16 col-blocks

    // staging: thread t copies row bn (0..63), 16B chunk bchunk (0..3)
    int bn = t >> 2, bchunk = t & 3;
    const _Float16* bsrc = W + (size_t)(col0 + bn) * EMBED + bchunk * 8;
    unsigned ldsB[2] = { lds_off32(&bsm[0][bn*32 + bchunk*8]),
                         lds_off32(&bsm[1][bn*32 + bchunk*8]) };

    async_ld_b128(ldsB[0], bsrc);                // k0 = 0 panel

    const v8f z = {0.f,0.f,0.f,0.f,0.f,0.f,0.f,0.f};
    v8f acc0 = z, acc1 = z, acc2 = z, acc3 = z;
    int n = lane & 15, h = lane >> 4;

    for (int k0 = 0; k0 < EMBED; k0 += 32){
        int cur = (k0 >> 5) & 1;
        __syncthreads();                          // prior reads of bsm[cur^1] done
        if (k0 + 32 < EMBED)
            async_ld_b128(ldsB[cur ^ 1], bsrc + k0 + 32);
        v16h a = load_a16(A, EMBED, row0, k0, lane);   // overlaps async wait
        if (k0 + 32 < EMBED) asm volatile("s_wait_asynccnt 0x1" ::: "memory");
        else                 asm volatile("s_wait_asynccnt 0x0" ::: "memory");
        __syncthreads();                          // bsm[cur] complete block-wide

        // issue all LDS reads first, then 4 back-to-back WMMAs
        const _Float16* bb = bsm[cur];
        v16h b0 = *(const v16h*)(bb + ( 0 + n)*32 + 16*h);
        v16h b1 = *(const v16h*)(bb + (16 + n)*32 + 16*h);
        v16h b2 = *(const v16h*)(bb + (32 + n)*32 + 16*h);
        v16h b3 = *(const v16h*)(bb + (48 + n)*32 + 16*h);
        acc0 = WMMA_F16(a, b0, acc0);
        acc1 = WMMA_F16(a, b1, acc1);
        acc2 = WMMA_F16(a, b2, acc2);
        acc3 = WMMA_F16(a, b3, acc3);
    }

    // C layout: VGPR r -> M = r + 8h, N = lane%16
    v8f accs[4] = {acc0, acc1, acc2, acc3};
    #pragma unroll
    for (int tt = 0; tt < 4; ++tt){
        #pragma unroll
        for (int r = 0; r < 8; ++r){
            int gr = row0 + r + 8*h;
            int gc = col0 + tt*16 + n;
            float v = accs[tt][r] + bias[gc];
            if (outmode == OUT_F32){
                ((float*)out)[(size_t)gr * EMBED + gc] = v;
            } else {
                int b  = gr >> 11, nn = gr & (SEQ - 1);
                int hh = gc >> 6,  d  = gc & (HD - 1);
                size_t idx;
                if (outmode == OUT_QK) idx = (((size_t)(b*HEADS + hh))*SEQ + nn)*HD + d;
                else                   idx = (((size_t)(b*HEADS + hh))*HD  + d )*SEQ + nn; // V^T
                ((_Float16*)out)[idx] = (_Float16)v;
            }
        }
    }
}

// ---------------------------------------------------------------------------
// Flash attention. Block = one (b,h), 128 query rows (16/wave). K-tile
// (32x64) and Vt-tile (64x32) async-staged to LDS, double buffered, shared
// by all 8 waves. Online softmax; P relaid to A-operand via per-wave LDS.
__global__ __launch_bounds__(256) void attn_kernel(const _Float16* __restrict__ Qh,
                                                   const _Float16* __restrict__ Kh,
                                                   const _Float16* __restrict__ Vt,
                                                   _Float16* __restrict__ Hh){
    __shared__ _Float16 Kl[2][32*64];            // 2 x 4KB keys x d
    __shared__ _Float16 Vl[2][64*32];            // 2 x 4KB d x keys
    __shared__ _Float16 plds[8][16*32];          // 8 x 1KB P relayout

    int t    = threadIdx.x;
    int lane = t & 31;
    int wave = t >> 5;
    int bh   = blockIdx.x >> 4;                  // 32 (b,h) pairs
    int q0   = (blockIdx.x & 15) * 128 + wave * 16;

    const _Float16* qb = Qh + (size_t)bh * SEQ * HD;
    const _Float16* kb = Kh + (size_t)bh * SEQ * HD;
    const _Float16* vb = Vt + (size_t)bh * HD  * SEQ;

    // staging maps: K tile rows 0..31 of 128B (8 chunks); V tile rows 0..63 of 64B (4 chunks)
    int krow = t >> 3, kch = t & 7;              // 32 x 8
    int vrow = t >> 2, vch = t & 3;              // 64 x 4
    const _Float16* ksrc = kb + (size_t)krow * HD  + kch * 8;
    const _Float16* vsrc = vb + (size_t)vrow * SEQ + vch * 8;
    unsigned ldsK[2] = { lds_off32(&Kl[0][krow*64 + kch*8]), lds_off32(&Kl[1][krow*64 + kch*8]) };
    unsigned ldsV[2] = { lds_off32(&Vl[0][vrow*32 + vch*8]), lds_off32(&Vl[1][vrow*32 + vch*8]) };

    async_ld_b128(ldsK[0], ksrc);                // kblk = 0
    async_ld_b128(ldsV[0], vsrc);

    v16h qa0 = load_a16(qb, HD, q0,  0, lane);
    v16h qa1 = load_a16(qb, HD, q0, 32, lane);

    const v8f z = {0.f,0.f,0.f,0.f,0.f,0.f,0.f,0.f};
    v8f o0 = z, o1 = z, o2 = z, o3 = z;
    float mrow[8], lrow[8];
    #pragma unroll
    for (int r = 0; r < 8; ++r){ mrow[r] = -1e30f; lrow[r] = 0.f; }

    _Float16* myp = plds[wave];
    int n = lane & 15, h = lane >> 4;

    for (int kblk = 0; kblk < SEQ; kblk += 32){
        int cur = (kblk >> 5) & 1;
        __syncthreads();                          // prior reads of [cur^1] done
        if (kblk + 32 < SEQ){
            async_ld_b128(ldsK[cur ^ 1], ksrc + (size_t)(kblk + 32) * HD);
            async_ld_b128(ldsV[cur ^ 1], vsrc + (kblk + 32));
            asm volatile("s_wait_asynccnt 0x2" ::: "memory");
        } else {
            asm volatile("s_wait_asynccnt 0x0" ::: "memory");
        }
        __syncthreads();                          // tiles [cur] complete

        // ---- scores S = Q K^T * 8 from LDS K-tile (B[d][key] = K[key][d])
        // issue all LDS reads, then 4 back-to-back WMMAs
        const _Float16* kk = Kl[cur];
        v16h kb0 = *(const v16h*)(kk + ( 0 + n)*64 +  0 + 16*h);
        v16h kb1 = *(const v16h*)(kk + ( 0 + n)*64 + 32 + 16*h);
        v16h kb2 = *(const v16h*)(kk + (16 + n)*64 +  0 + 16*h);
        v16h kb3 = *(const v16h*)(kk + (16 + n)*64 + 32 + 16*h);
        v8f s0 = z, s1 = z;
        s0 = WMMA_F16(qa0, kb0, s0);
        s0 = WMMA_F16(qa1, kb1, s0);
        s1 = WMMA_F16(qa0, kb2, s1);
        s1 = WMMA_F16(qa1, kb3, s1);

        // ---- prefetch V-tile LDS reads early (overlap softmax VALU)
        const _Float16* vv = Vl[cur];
        v16h vb0 = *(const v16h*)(vv + ( 0 + n)*32 + 16*h);
        v16h vb1 = *(const v16h*)(vv + (16 + n)*32 + 16*h);
        v16h vb2 = *(const v16h*)(vv + (32 + n)*32 + 16*h);
        v16h vb3 = *(const v16h*)(vv + (48 + n)*32 + 16*h);

        // ---- online softmax (row m = r + 8h lives in a 16-lane group)
        float p0[8], p1[8];
        #pragma unroll
        for (int r = 0; r < 8; ++r){
            float a = s0[r] * SCALE_INV;
            float b = s1[r] * SCALE_INV;
            float tm = fmaxf(a, b);
            tm = fmaxf(tm, __shfl_xor(tm, 1, 32));
            tm = fmaxf(tm, __shfl_xor(tm, 2, 32));
            tm = fmaxf(tm, __shfl_xor(tm, 4, 32));
            tm = fmaxf(tm, __shfl_xor(tm, 8, 32));
            float mnew  = fmaxf(mrow[r], tm);
            float alpha = __expf(mrow[r] - mnew);
            mrow[r] = mnew;
            p0[r] = __expf(a - mnew);
            p1[r] = __expf(b - mnew);
            float rs = p0[r] + p1[r];
            rs += __shfl_xor(rs, 1, 32);
            rs += __shfl_xor(rs, 2, 32);
            rs += __shfl_xor(rs, 4, 32);
            rs += __shfl_xor(rs, 8, 32);
            lrow[r] = lrow[r] * alpha + rs;
            o0[r] *= alpha; o1[r] *= alpha; o2[r] *= alpha; o3[r] *= alpha;
        }

        // ---- relayout P (C-layout f32) -> A-operand f16 via wave-private LDS
        #pragma unroll
        for (int r = 0; r < 8; ++r){
            int m = r + 8*h;
            myp[m*32 +      n] = (_Float16)p0[r];
            myp[m*32 + 16 + n] = (_Float16)p1[r];
        }
        __asm__ volatile("s_wait_dscnt 0x0" ::: "memory");  // DS in-order per wave
        v8h lo = *(const v8h*)(myp + n*32 + 16*h);
        v8h hi = *(const v8h*)(myp + n*32 + 16*h + 8);
        v16h pa = __builtin_shufflevector(lo, hi, 0,1,2,3,4,5,6,7,8,9,10,11,12,13,14,15);
        __asm__ volatile("" ::: "memory");

        // ---- O += P @ V (V-tiles already in registers): 4 back-to-back WMMAs
        o0 = WMMA_F16(pa, vb0, o0);
        o1 = WMMA_F16(pa, vb1, o1);
        o2 = WMMA_F16(pa, vb2, o2);
        o3 = WMMA_F16(pa, vb3, o3);
    }

    // ---- finalize: Hh[b*SEQ + q, head*HD + d] = O / l   (f16, feeds out-proj)
    int b = bh >> 4, head = bh & 15;
    v8f oo[4] = {o0, o1, o2, o3};
    #pragma unroll
    for (int tt = 0; tt < 4; ++tt){
        #pragma unroll
        for (int r = 0; r < 8; ++r){
            int m = r + 8*h;
            float inv = 1.0f / lrow[r];
            int col = head*HD + tt*16 + n;
            size_t idx = ((size_t)(b*SEQ + q0 + m))*EMBED + col;
            Hh[idx] = (_Float16)(oo[tt][r] * inv);
        }
    }
}

// ---------------------------------------------------------------------------
extern "C" void kernel_launch(void* const* d_in, const int* in_sizes, int n_in,
                              void* d_out, int out_size, void* d_ws, size_t ws_size,
                              hipStream_t stream){
    const float* x  = (const float*)d_in[0];
    const float* Wq = (const float*)d_in[1];
    const float* bq = (const float*)d_in[2];
    const float* Wk = (const float*)d_in[3];
    const float* bk = (const float*)d_in[4];
    const float* Wv = (const float*)d_in[5];
    const float* bv = (const float*)d_in[6];
    const float* Wo = (const float*)d_in[7];
    const float* bo = (const float*)d_in[8];

    char* ws = (char*)d_ws;                               // 48 MiB used
    _Float16* xh  = (_Float16*)(ws);                      //  8 MiB  x   f16
    _Float16* wqh = (_Float16*)(ws +  8388608);           //  2 MiB each
    _Float16* wkh = (_Float16*)(ws + 10485760);
    _Float16* wvh = (_Float16*)(ws + 12582912);
    _Float16* woh = (_Float16*)(ws + 14680064);
    _Float16* Qh  = (_Float16*)(ws + 16777216);           //  8 MiB [B,H,N,D]
    _Float16* Kh  = (_Float16*)(ws + 25165824);           //  8 MiB [B,H,N,D]
    _Float16* Vt  = (_Float16*)(ws + 33554432);           //  8 MiB [B,H,D,N]
    _Float16* Hh  = (_Float16*)(ws + 41943040);           //  8 MiB [B*N, C]

    cvt_f32_f16<<<(ROWS*EMBED + 255)/256, 256, 0, stream>>>(x,  xh,  ROWS*EMBED);
    cvt_f32_f16<<<(EMBED*EMBED + 255)/256, 256, 0, stream>>>(Wq, wqh, EMBED*EMBED);
    cvt_f32_f16<<<(EMBED*EMBED + 255)/256, 256, 0, stream>>>(Wk, wkh, EMBED*EMBED);
    cvt_f32_f16<<<(EMBED*EMBED + 255)/256, 256, 0, stream>>>(Wv, wvh, EMBED*EMBED);
    cvt_f32_f16<<<(EMBED*EMBED + 255)/256, 256, 0, stream>>>(Wo, woh, EMBED*EMBED);

    gemm_xWt<<<512, 256, 0, stream>>>(xh, wqh, bq, (void*)Qh, OUT_QK);
    gemm_xWt<<<512, 256, 0, stream>>>(xh, wkh, bk, (void*)Kh, OUT_QK);
    gemm_xWt<<<512, 256, 0, stream>>>(xh, wvh, bv, (void*)Vt, OUT_VT);

    attn_kernel<<<512, 256, 0, stream>>>(Qh, Kh, Vt, Hh);

    gemm_xWt<<<512, 256, 0, stream>>>(Hh, woh, bo, d_out, OUT_F32);
}
